// BigramLanguageModel_57105885168154
// MI455X (gfx1250) — compile-verified, matrix operand order checked by reading, not run on hardware
//
#include <hip/hip_runtime.h>
#include <hip/hip_bf16.h>

// ---------------------------------------------------------------------------
// GPT-style transformer forward for MI455X (gfx1250), wave32 + WMMA.
// Precision: fp16 WMMA inputs, fp32 accumulation (v_wmma_f32_16x16x32_f16).
// Weights converted to fp16 once per launch (~23MB -> resident in 192MB L2),
// so every GEMM after first touch is compute-bound on the WMMA pipe.
// GEMM uses double-buffered LDS with the gfx1250 async global->LDS datapath
// (global_load_async_to_lds_b128 / s_wait_asynccnt) for the A tile.
// ---------------------------------------------------------------------------

typedef __attribute__((ext_vector_type(16))) _Float16 v16h;
typedef __attribute__((ext_vector_type(8)))  _Float16 v8h;
typedef __attribute__((ext_vector_type(8)))  float    v8f;

#define B_  2
#define T_  2048
#define E_  256
#define H_  8
#define L_  4
#define V_  32000
#define HD_ 32
#define BT_ (B_ * T_)

static __device__ __forceinline__ v8f wmma16(v16h a, v16h b, v8f c) {
  // (neg_a, A, neg_b, B, c_mod, C, reuse_a, reuse_b)
  return __builtin_amdgcn_wmma_f32_16x16x32_f16(false, a, false, b,
                                                (short)0, c, false, false);
}

// ---------------------------------------------------------------------------
// fp32 -> fp16 conversion (weights); grid-stride.
// ---------------------------------------------------------------------------
__global__ void k_cvt(const float* __restrict__ s, _Float16* __restrict__ d,
                      long n) {
  long i = (long)blockIdx.x * blockDim.x + threadIdx.x;
  long stride = (long)gridDim.x * blockDim.x;
  for (; i < n; i += stride) d[i] = (_Float16)s[i];
}

// ---------------------------------------------------------------------------
// x[b,t,:] = tok_emb[X[b,t],:] + pos_emb[t,:]
// ---------------------------------------------------------------------------
__global__ __launch_bounds__(256)
void k_embed(const int* __restrict__ X, const float* __restrict__ tok,
             const float* __restrict__ pos, float* __restrict__ x) {
  int bt = blockIdx.x;
  int e  = threadIdx.x;
  int t  = bt % T_;
  int id = X[bt];
  x[(long)bt * E_ + e] = tok[(long)id * E_ + e] + pos[(long)t * E_ + e];
}

// ---------------------------------------------------------------------------
// LayerNorm over E=256, one 256-thread block (8 waves) per row, fp16 out.
// ---------------------------------------------------------------------------
__global__ __launch_bounds__(256)
void k_ln(const float* __restrict__ x, const float* __restrict__ g,
          const float* __restrict__ bia, _Float16* __restrict__ out) {
  __shared__ float red[256];
  int row = blockIdx.x, tid = threadIdx.x;
  float v = x[(long)row * E_ + tid];
  red[tid] = v;
  __syncthreads();
  for (int s = 128; s > 0; s >>= 1) {
    if (tid < s) red[tid] += red[tid + s];
    __syncthreads();
  }
  float mean = red[0] * (1.0f / E_);
  __syncthreads();
  float d = v - mean;
  red[tid] = d * d;
  __syncthreads();
  for (int s = 128; s > 0; s >>= 1) {
    if (tid < s) red[tid] += red[tid + s];
    __syncthreads();
  }
  float var = red[0] * (1.0f / E_);
  float r = rsqrtf(var + 1e-5f);
  out[(long)row * E_ + tid] = (_Float16)(d * r * g[tid] + bia[tid]);
}

// ---------------------------------------------------------------------------
// Generic batched fp16 WMMA GEMM: C = A[MxK] * B[KxN] (+bias)(+res)(ReLU)
// Block tile 128x128, BK=32, double-buffered LDS, one barrier per k-step.
// 8 waves as 4x2; each wave computes 32x64 as a 2x4 grid of
// v_wmma_f32_16x16x32_f16 tiles (8 WMMAs per 6 fragment ds_load pairs).
// A tile staged via async global->LDS (ASYNCcnt); B staged transposed via
// registers so both fragments read as two contiguous 16B chunks per lane
// (ISA 16-bit A layout: e<8 -> k=e+8h, e>=8 -> k=e+8+8h).
// Requires M % 128 == 0 and K % 32 == 0 (true for all call sites).
// Batch z: A offset = (z / zdivA)*sAz, B offset = (z % zmodB)*sBz.
// ---------------------------------------------------------------------------
#define BM 128
#define BN 128
#define BK 32

__global__ __launch_bounds__(256)
void k_gemm(const _Float16* __restrict__ A, int lda, long sAz, int zdivA,
            const _Float16* __restrict__ B, int ldb, long sBz, int zmodB,
            float* __restrict__ outF, _Float16* __restrict__ outH,
            int ldc, long sCz,
            const float* __restrict__ bias, const float* __restrict__ resF,
            int M, int N, int K, int relu) {
  __shared__ __align__(16) _Float16 As[2][BM][BK];  // 2 x 8 KB
  __shared__ __align__(16) _Float16 Bs[2][BN][BK];  // 2 x 8 KB, [n][k]
  const int tid  = threadIdx.x;
  const int lane = tid & 31;
  const int wid  = tid >> 5;
  const int li   = lane & 15;
  const int hh   = lane >> 4;
  const int m0   = blockIdx.y * BM;
  const int n0   = blockIdx.x * BN;
  const int bz   = blockIdx.z;
  const _Float16* Ab = A + (long)(bz / zdivA) * sAz + (long)m0 * lda;
  const _Float16* Bb = B + (long)(bz % zmodB) * sBz + n0;
  const long cb = (long)bz * sCz;
  const int wm = (wid & 3) * 32;   // wave row offset in tile
  const int wn = (wid >> 2) * 64;  // wave col offset in tile

  // A tile: 512 chunks of 8 halfs (16B), 2 per thread, async to LDS.
  auto stageA = [&](int buf, int k0) {
#pragma unroll
    for (int rr = 0; rr < 2; ++rr) {
      int idx = tid + rr * 256;
      int i = idx >> 2;
      int j = (idx & 3) * 8;
      unsigned lds = (unsigned)(size_t)&As[buf][i][j];
      const _Float16* g = Ab + (long)i * lda + k0 + j;
      asm volatile("global_load_async_to_lds_b128 %0, %1, off"
                   :: "v"(lds), "v"(g) : "memory");
    }
  };

  // B tile: 512 chunks of 8 halfs along n, 2 per thread, via registers
  // (transposed into Bs[n][k]); zero-filled past N.
  _Float16 tb[2][8];
  auto loadB = [&](int k0) {
#pragma unroll
    for (int rr = 0; rr < 2; ++rr) {
      int idx = tid + rr * 256;
      int kk = idx >> 4;
      int jc = (idx & 15) * 8;
      if (n0 + jc + 8 <= N) {
        *(uint4*)tb[rr] = *(const uint4*)(Bb + (long)(k0 + kk) * ldb + jc);
      } else {
#pragma unroll
        for (int c = 0; c < 8; ++c) tb[rr][c] = (_Float16)0.f;
      }
    }
  };
  auto storeB = [&](int buf) {
#pragma unroll
    for (int rr = 0; rr < 2; ++rr) {
      int idx = tid + rr * 256;
      int kk = idx >> 4;
      int jc = (idx & 15) * 8;
#pragma unroll
      for (int c = 0; c < 8; ++c) Bs[buf][jc + c][kk] = tb[rr][c];
    }
  };

  v8f acc[2][4] = {};

  stageA(0, 0);
  loadB(0);
  storeB(0);
  asm volatile("s_wait_asynccnt 0x0" ::: "memory");
  __syncthreads();

  const int nIter = K / BK;
  for (int it = 0; it < nIter; ++it) {
    const int cur = it & 1, nxt = cur ^ 1;
    const bool hasNext = (it + 1) < nIter;
    if (hasNext) {                 // prefetch next tile while computing
      stageA(nxt, (it + 1) * BK);
      loadB((it + 1) * BK);
    }

    v16h af[2], bf[4];
#pragma unroll
    for (int mt = 0; mt < 2; ++mt) {
      int row = wm + mt * 16 + li;
      *(v8h*)&af[mt]         = *(const v8h*)&As[cur][row][hh * 8];
      *(((v8h*)&af[mt]) + 1) = *(const v8h*)&As[cur][row][16 + hh * 8];
    }
#pragma unroll
    for (int nt = 0; nt < 4; ++nt) {
      int col = wn + nt * 16 + li;
      *(v8h*)&bf[nt]         = *(const v8h*)&Bs[cur][col][hh * 8];
      *(((v8h*)&bf[nt]) + 1) = *(const v8h*)&Bs[cur][col][16 + hh * 8];
    }
#pragma unroll
    for (int mt = 0; mt < 2; ++mt)
#pragma unroll
      for (int nt = 0; nt < 4; ++nt)
        acc[mt][nt] = wmma16(af[mt], bf[nt], acc[mt][nt]);

    if (hasNext) storeB(nxt);
    asm volatile("s_wait_asynccnt 0x0" ::: "memory");
    __syncthreads();               // single barrier per k-step (ping-pong)
  }

  // ---- epilogue: C layout lane=col(li), VGPR r -> row r+8*hh ----
#pragma unroll
  for (int mt = 0; mt < 2; ++mt) {
#pragma unroll
    for (int nt = 0; nt < 4; ++nt) {
      int col = n0 + wn + nt * 16 + li;
#pragma unroll
      for (int r = 0; r < 8; ++r) {
        int row = m0 + wm + mt * 16 + r + 8 * hh;
        if (row < M && col < N) {
          float v = acc[mt][nt][r];
          if (bias) v += bias[col];
          long o = cb + (long)row * ldc + col;
          if (resF) v += resF[o];
          if (relu) v = fmaxf(v, 0.f);
          if (outF) outF[o] = v;
          if (outH) outH[o] = (_Float16)v;
        }
      }
    }
  }
}

// ---------------------------------------------------------------------------
// Flash-attention: one wave per (b, h, 16-query tile). HD=32 => S tile is a
// single v_wmma_f32_16x16x32_f16 per 16-key group. Online softmax with
// __shfl_xor row reductions over the 16-lane halves; P re-staged through LDS
// to re-enter in A-fragment layout for the P*V WMMAs.
// ---------------------------------------------------------------------------
__global__ __launch_bounds__(32)
void k_attn(const _Float16* __restrict__ Q, const _Float16* __restrict__ Kh,
            const _Float16* __restrict__ Vh, _Float16* __restrict__ O) {
  __shared__ __align__(16) _Float16 Ps[16][32];
  const int lane = threadIdx.x;
  const int li = lane & 15;
  const int hh = lane >> 4;
  const int qt = blockIdx.x;
  const int h  = blockIdx.y;
  const int b  = blockIdx.z;
  const long base = ((long)(b * H_ + h)) * T_ * HD_;
  const _Float16* Qp = Q + base;
  const _Float16* Kp = Kh + base;
  const _Float16* Vp = Vh + base;
  const int q0 = qt * 16;
  const float scale = 0.0625f;  // E^-0.5 (reference scales by full E)

  v16h qa;
  *(v8h*)&qa         = *(const v8h*)(Qp + (long)(q0 + li) * HD_ + hh * 8);
  *(((v8h*)&qa) + 1) = *(const v8h*)(Qp + (long)(q0 + li) * HD_ + 16 + hh * 8);

  float m[8], l[8];
#pragma unroll
  for (int r = 0; r < 8; ++r) { m[r] = -1e30f; l[r] = 0.f; }
  v8f o0 = {}, o1 = {};

  for (int kb = 0; kb < q0 + 16; kb += 32) {
    // K^T fragments: lane = key column -> contiguous along HD
    v16h kf0, kf1;
    *(v8h*)&kf0         = *(const v8h*)(Kp + (long)(kb + li) * HD_ + hh * 8);
    *(((v8h*)&kf0) + 1) = *(const v8h*)(Kp + (long)(kb + li) * HD_ + 16 + hh * 8);
    *(v8h*)&kf1         = *(const v8h*)(Kp + (long)(kb + 16 + li) * HD_ + hh * 8);
    *(((v8h*)&kf1) + 1) = *(const v8h*)(Kp + (long)(kb + 16 + li) * HD_ + 16 + hh * 8);
    v8f s0 = {}, s1 = {};
    s0 = wmma16(qa, kf0, s0);
    s1 = wmma16(qa, kf1, s1);

#pragma unroll
    for (int r = 0; r < 8; ++r) {
      int t = q0 + r + 8 * hh;
      float f0 = s0[r] * scale;
      float f1 = s1[r] * scale;
      if (kb + li > t)      f0 = -1e30f;   // causal mask
      if (kb + 16 + li > t) f1 = -1e30f;
      float mrow = fmaxf(f0, f1);
      mrow = fmaxf(mrow, __shfl_xor(mrow, 1));
      mrow = fmaxf(mrow, __shfl_xor(mrow, 2));
      mrow = fmaxf(mrow, __shfl_xor(mrow, 4));
      mrow = fmaxf(mrow, __shfl_xor(mrow, 8));
      float mn = fmaxf(m[r], mrow);
      float al = __expf(m[r] - mn);
      float p0 = __expf(f0 - mn);
      float p1 = __expf(f1 - mn);
      float ps = p0 + p1;
      ps += __shfl_xor(ps, 1);
      ps += __shfl_xor(ps, 2);
      ps += __shfl_xor(ps, 4);
      ps += __shfl_xor(ps, 8);
      l[r] = l[r] * al + ps;
      m[r] = mn;
      o0[r] *= al;
      o1[r] *= al;
      Ps[r + 8 * hh][li]      = (_Float16)p0;
      Ps[r + 8 * hh][16 + li] = (_Float16)p1;
    }
    __syncthreads();

    // P back as A-fragment (rows = queries, K-dim = 32 keys)
    v16h pa;
    *(v8h*)&pa         = *(const v8h*)&Ps[li][hh * 8];
    *(((v8h*)&pa) + 1) = *(const v8h*)&Ps[li][16 + hh * 8];

    // V fragments (K-dim strided over rows of [T,HD] -> scalar gathers)
    v16h vf0, vf1;
#pragma unroll
    for (int e = 0; e < 16; ++e) {
      int kk = e + 8 * hh + (e < 8 ? 0 : 8);
      vf0[e] = Vp[(long)(kb + kk) * HD_ + li];
      vf1[e] = Vp[(long)(kb + kk) * HD_ + 16 + li];
    }
    o0 = wmma16(pa, vf0, o0);
    o1 = wmma16(pa, vf1, o1);
    __syncthreads();
  }

#pragma unroll
  for (int r = 0; r < 8; ++r) {
    float inv = 1.0f / l[r];
    int t = q0 + r + 8 * hh;
    long ob = ((long)b * T_ + t) * E_ + h * HD_;   // head-concat layout
    O[ob + li]      = (_Float16)(o0[r] * inv);
    O[ob + 16 + li] = (_Float16)(o1[r] * inv);
  }
}

// ---------------------------------------------------------------------------
// Host orchestration
// ---------------------------------------------------------------------------
extern "C" void kernel_launch(void* const* d_in, const int* in_sizes, int n_in,
                              void* d_out, int out_size, void* d_ws,
                              size_t ws_size, hipStream_t stream) {
  (void)in_sizes; (void)n_in; (void)out_size; (void)ws_size;
  const int*   X    = (const int*)  d_in[0];
  const float* tok  = (const float*)d_in[1];
  const float* pos  = (const float*)d_in[2];
  const float* wq   = (const float*)d_in[3];
  const float* wk   = (const float*)d_in[4];
  const float* wv   = (const float*)d_in[5];
  const float* wo   = (const float*)d_in[6];
  const float* bo   = (const float*)d_in[7];
  const float* w1   = (const float*)d_in[8];
  const float* b1   = (const float*)d_in[9];
  const float* w2   = (const float*)d_in[10];
  const float* b2   = (const float*)d_in[11];
  const float* ln1g = (const float*)d_in[12];
  const float* ln1b = (const float*)d_in[13];
  const float* ln2g = (const float*)d_in[14];
  const float* ln2b = (const float*)d_in[15];
  const float* lnfg = (const float*)d_in[16];
  const float* lnfb = (const float*)d_in[17];
  const float* wh   = (const float*)d_in[18];
  const float* bh   = (const float*)d_in[19];

  char* ws = (char*)d_ws;
  size_t off = 0;
  auto alloc = [&](size_t bytes) -> void* {
    off = (off + 255) & ~(size_t)255;
    void* p = ws + off;
    off += bytes;
    return p;
  };

  float*    xf   = (float*)   alloc((size_t)BT_ * E_ * 4);
  _Float16* hln  = (_Float16*)alloc((size_t)BT_ * E_ * 2);
  _Float16* qh   = (_Float16*)alloc((size_t)BT_ * E_ * 2);
  _Float16* kh   = (_Float16*)alloc((size_t)BT_ * E_ * 2);
  _Float16* vh   = (_Float16*)alloc((size_t)BT_ * E_ * 2);
  _Float16* ah   = (_Float16*)alloc((size_t)BT_ * E_ * 2);
  _Float16* h1   = (_Float16*)alloc((size_t)BT_ * 4 * E_ * 2);
  _Float16* xfin = (_Float16*)alloc((size_t)BT_ * E_ * 2);
  _Float16* wqh  = (_Float16*)alloc((size_t)L_ * H_ * E_ * HD_ * 2);
  _Float16* wkh  = (_Float16*)alloc((size_t)L_ * H_ * E_ * HD_ * 2);
  _Float16* wvh  = (_Float16*)alloc((size_t)L_ * H_ * E_ * HD_ * 2);
  _Float16* woh  = (_Float16*)alloc((size_t)L_ * E_ * E_ * 2);
  _Float16* w1h  = (_Float16*)alloc((size_t)L_ * E_ * 4 * E_ * 2);
  _Float16* w2h  = (_Float16*)alloc((size_t)L_ * 4 * E_ * E_ * 2);
  _Float16* whh  = (_Float16*)alloc((size_t)E_ * V_ * 2);

  // weight conversion (L2-resident afterwards)
  k_cvt<<<1024, 256, 0, stream>>>(wq, wqh, (long)L_ * H_ * E_ * HD_);
  k_cvt<<<1024, 256, 0, stream>>>(wk, wkh, (long)L_ * H_ * E_ * HD_);
  k_cvt<<<1024, 256, 0, stream>>>(wv, wvh, (long)L_ * H_ * E_ * HD_);
  k_cvt<<<1024, 256, 0, stream>>>(wo, woh, (long)L_ * E_ * E_);
  k_cvt<<<1024, 256, 0, stream>>>(w1, w1h, (long)L_ * E_ * 4 * E_);
  k_cvt<<<1024, 256, 0, stream>>>(w2, w2h, (long)L_ * 4 * E_ * E_);
  k_cvt<<<2048, 256, 0, stream>>>(wh, whh, (long)E_ * V_);

  k_embed<<<BT_, 256, 0, stream>>>(X, tok, pos, xf);

  for (int l = 0; l < L_; ++l) {
    k_ln<<<BT_, 256, 0, stream>>>(xf, ln1g + l * E_, ln1b + l * E_, hln);

    // QKV: batched over z = b*H + h; A depends on b (z/H), B on h (z%H)
    dim3 gq((HD_ + BN - 1) / BN, (T_ + BM - 1) / BM, B_ * H_);
    k_gemm<<<gq, 256, 0, stream>>>(hln, E_, (long)T_ * E_, H_,
                                   wqh + (long)l * H_ * E_ * HD_, HD_,
                                   (long)E_ * HD_, H_,
                                   nullptr, qh, HD_, (long)T_ * HD_,
                                   nullptr, nullptr, T_, HD_, E_, 0);
    k_gemm<<<gq, 256, 0, stream>>>(hln, E_, (long)T_ * E_, H_,
                                   wkh + (long)l * H_ * E_ * HD_, HD_,
                                   (long)E_ * HD_, H_,
                                   nullptr, kh, HD_, (long)T_ * HD_,
                                   nullptr, nullptr, T_, HD_, E_, 0);
    k_gemm<<<gq, 256, 0, stream>>>(hln, E_, (long)T_ * E_, H_,
                                   wvh + (long)l * H_ * E_ * HD_, HD_,
                                   (long)E_ * HD_, H_,
                                   nullptr, vh, HD_, (long)T_ * HD_,
                                   nullptr, nullptr, T_, HD_, E_, 0);

    dim3 ga(T_ / 16, H_, B_);
    k_attn<<<ga, 32, 0, stream>>>(qh, kh, vh, ah);

    // x = x + attn @ wo + bo
    dim3 go((E_ + BN - 1) / BN, BT_ / BM, 1);
    k_gemm<<<go, 256, 0, stream>>>(ah, E_, 0, 1,
                                   woh + (long)l * E_ * E_, E_, 0, 1,
                                   xf, nullptr, E_, 0,
                                   bo + l * E_, xf, BT_, E_, E_, 0);

    k_ln<<<BT_, 256, 0, stream>>>(xf, ln2g + l * E_, ln2b + l * E_, hln);

    // h1 = relu(hln @ w1 + b1)
    dim3 g1((4 * E_ + BN - 1) / BN, BT_ / BM, 1);
    k_gemm<<<g1, 256, 0, stream>>>(hln, E_, 0, 1,
                                   w1h + (long)l * E_ * 4 * E_, 4 * E_, 0, 1,
                                   nullptr, h1, 4 * E_, 0,
                                   b1 + l * 4 * E_, nullptr,
                                   BT_, 4 * E_, E_, 1);

    // x = x + h1 @ w2 + b2
    dim3 g2((E_ + BN - 1) / BN, BT_ / BM, 1);
    k_gemm<<<g2, 256, 0, stream>>>(h1, 4 * E_, 0, 1,
                                   w2h + (long)l * 4 * E_ * E_, E_, 0, 1,
                                   xf, nullptr, E_, 0,
                                   b2 + l * E_, xf, BT_, E_, 4 * E_, 0);
  }

  k_ln<<<BT_, 256, 0, stream>>>(xf, lnfg, lnfb, xfin);

  // logits = xfin @ w_head + b_head  (dominant GEMM: 4096x32000x256)
  dim3 gh(V_ / BN, BT_ / BM, 1);
  k_gemm<<<gh, 256, 0, stream>>>(xfin, E_, 0, 1,
                                 whh, V_, 0, 1,
                                 (float*)d_out, nullptr, V_, 0,
                                 bh, nullptr, BT_, V_, E_, 0);
}
